// TSAKT_78932908965946
// MI455X (gfx1250) — matrix operand order repaired and checked by reading, not compiled, not run on hardware
//
#include <hip/hip_runtime.h>
#include <hip/hip_bf16.h>
#include <stdint.h>

// Problem constants (match reference)
#define BB 32
#define SS 512
#define EEc 512
#define HH 8
#define DHH 64
#define LLn 2
static constexpr float SCALEv = 0.125f;          // 1/sqrt(64)
static constexpr float NEGV   = -1000000000.0f;

typedef __attribute__((ext_vector_type(16))) __bf16 v16bf;
typedef __attribute__((ext_vector_type(8)))  float  v8f;
typedef int v4i_vs __attribute__((vector_size(16)));   // matches builtin param type

union Frag16 { uint4 q[2]; v16bf v; __bf16 h[16]; };

__device__ __forceinline__ v8f wmma_bf16(v16bf a, v16bf b, v8f c) {
  // D = A(16x32 bf16) x B(32x16 bf16) + C(16x16 f32)
  return __builtin_amdgcn_wmma_f32_16x16x32_bf16(false, a, false, b, (short)0, c,
                                                 false, false);
}

// ---- gfx1250 async global->LDS path (guarded; falls back to sync copy) ----
#ifndef __has_builtin
#define __has_builtin(x) 0
#endif
#if __has_builtin(__builtin_amdgcn_global_load_async_to_lds_b128) && \
    __has_builtin(__builtin_amdgcn_s_wait_asynccnt)
#define HAVE_ASYNC_LDS 1
#else
#define HAVE_ASYNC_LDS 0
#endif

__device__ __forceinline__ void async_cp16(__bf16* lds_dst, const __bf16* gsrc) {
#if HAVE_ASYNC_LDS
  __builtin_amdgcn_global_load_async_to_lds_b128(
      (__attribute__((address_space(1))) v4i_vs*)gsrc,
      (__attribute__((address_space(3))) v4i_vs*)lds_dst, 0, 0);
#else
  *(uint4*)lds_dst = *(const uint4*)gsrc;
#endif
}
__device__ __forceinline__ void async_join() {
#if HAVE_ASYNC_LDS
  __builtin_amdgcn_s_wait_asynccnt(0);
#endif
}

// ---------------------------------------------------------------------------
// fp32 -> bf16 weight conversion
// ---------------------------------------------------------------------------
__global__ void cvt_bf16_kernel(const float* __restrict__ src,
                                __bf16* __restrict__ dst, int n) {
  int i = blockIdx.x * 256 + threadIdx.x;
  if (i < n) dst[i] = (__bf16)src[i];
}

// ---------------------------------------------------------------------------
// Embedding gather + label-gated concat -> inputs [B,S,2E] bf16, query bf16
// grid = B*S blocks, 256 threads (E/2 = 256)
// ---------------------------------------------------------------------------
__global__ __launch_bounds__(256)
void embed_kernel(const int* __restrict__ item_inputs, const int* __restrict__ skill_inputs,
                  const int* __restrict__ label_inputs, const int* __restrict__ item_ids,
                  const int* __restrict__ skill_ids,
                  const float* __restrict__ item_emb, const float* __restrict__ skill_emb,
                  __bf16* __restrict__ inp, __bf16* __restrict__ qry) {
  int bs = blockIdx.x;
  int t  = threadIdx.x;               // 0..255
  float lab = (float)label_inputs[bs];
  int ii = item_inputs[bs], si = skill_inputs[bs];
  float ie = item_emb[(size_t)ii * 256 + t];
  float se = skill_emb[(size_t)si * 256 + t];
  __bf16* ip = inp + (size_t)bs * (2 * EEc);
  ip[t]             = (__bf16)(ie * lab);
  ip[256 + t]       = (__bf16)(se * lab);
  ip[EEc + t]       = (__bf16)(ie * (1.0f - lab));
  ip[EEc + 256 + t] = (__bf16)(se * (1.0f - lab));
  int qi = item_ids[bs], qs = skill_ids[bs];
  __bf16* qp = qry + (size_t)bs * EEc;
  qp[t]       = (__bf16)item_emb[(size_t)qi * 256 + t];
  qp[256 + t] = (__bf16)skill_emb[(size_t)qs * 256 + t];
}

// ---------------------------------------------------------------------------
// bf16 WMMA GEMM: C[M,N] = act(A[M,K] @ W[K,N] + bias)
// 256 threads / 8 waves; block tile 128x128; wave tile 32x64; K-step 32.
// Double-buffered LDS: A streamed via async global->LDS; W staged transposed
// (Bs[col][k], stride 40 -> conflict-free b128 fragment loads).
// Requires M%128==0, N%128==0, K%32==0.
// ---------------------------------------------------------------------------
#define ASTR 40
#define BSTR 40

__global__ __launch_bounds__(256)
void gemm_bf16_kernel(const __bf16* __restrict__ A, const __bf16* __restrict__ W,
                      const float* __restrict__ bias, __bf16* __restrict__ C,
                      int M, int N, int K, int do_relu) {
  __shared__ __bf16 As[2][128 * ASTR];
  __shared__ __bf16 Bs[2][128 * BSTR];

  const int tid  = threadIdx.x;
  const int wave = tid >> 5;
  const int lane = tid & 31;
  const int hs   = lane >> 4;         // half-wave select
  const int l16  = lane & 15;
  const int bm = blockIdx.x * 128;
  const int bn = blockIdx.y * 128;
  const int wm = (wave & 3) * 32;
  const int wn = (wave >> 2) * 64;

  // staging roles
  const int arow = tid >> 1, acol = (tid & 1) * 16;     // A tile 128x32
  const int bkrow = tid >> 3, bcol0 = (tid & 7) * 16;   // W tile 32x128

  const int T = K / 32;
  v8f acc[2][4] = {};
  Frag16 breg;

  // prologue: stage tile 0
  {
    const __bf16* bsrc = W + (size_t)bkrow * N + bn + bcol0;
    breg.q[0] = *(const uint4*)bsrc;
    breg.q[1] = *(const uint4*)(bsrc + 8);
    const __bf16* asrc = A + (size_t)(bm + arow) * K + acol;
    __bf16* adst = &As[0][arow * ASTR + acol];
    async_cp16(adst, asrc);
    async_cp16(adst + 8, asrc + 8);
    async_join();
#pragma unroll
    for (int i = 0; i < 16; ++i) Bs[0][(bcol0 + i) * BSTR + bkrow] = breg.h[i];
  }
  __syncthreads();

  for (int t = 0; t < T; ++t) {
    const int cur = t & 1, nxt = cur ^ 1;
    if (t + 1 < T) {
      // prefetch next W tile to registers; stream next A tile via async unit
      const __bf16* bsrc = W + (size_t)((t + 1) * 32 + bkrow) * N + bn + bcol0;
      breg.q[0] = *(const uint4*)bsrc;
      breg.q[1] = *(const uint4*)(bsrc + 8);
      const __bf16* asrc = A + (size_t)(bm + arow) * K + (t + 1) * 32 + acol;
      __bf16* adst = &As[nxt][arow * ASTR + acol];
      async_cp16(adst, asrc);
      async_cp16(adst + 8, asrc + 8);
    }

    // fragments (all contiguous b128 pairs) + 8 WMMAs
    Frag16 fa[2], fb[4];
#pragma unroll
    for (int f = 0; f < 2; ++f) {
      const __bf16* ap = &As[cur][(wm + f * 16 + l16) * ASTR + hs * 8];
      fa[f].q[0] = *(const uint4*)ap;
      fa[f].q[1] = *(const uint4*)(ap + 16);
    }
#pragma unroll
    for (int g = 0; g < 4; ++g) {
      const __bf16* bp = &Bs[cur][(wn + g * 16 + l16) * BSTR + hs * 8];
      fb[g].q[0] = *(const uint4*)bp;
      fb[g].q[1] = *(const uint4*)(bp + 16);
    }
#pragma unroll
    for (int f = 0; f < 2; ++f)
#pragma unroll
      for (int g = 0; g < 4; ++g) acc[f][g] = wmma_bf16(fa[f].v, fb[g].v, acc[f][g]);

    if (t + 1 < T) {
      async_join();
#pragma unroll
      for (int i = 0; i < 16; ++i) Bs[nxt][(bcol0 + i) * BSTR + bkrow] = breg.h[i];
      __syncthreads();
    }
  }

  // epilogue: bias + (relu) + bf16 store
#pragma unroll
  for (int f = 0; f < 2; ++f) {
#pragma unroll
    for (int g = 0; g < 4; ++g) {
      int col = bn + wn + g * 16 + l16;
      float bvv = bias ? bias[col] : 0.0f;
      int mbase = bm + wm + f * 16 + hs * 8;
#pragma unroll
      for (int r = 0; r < 8; ++r) {
        float v = acc[f][g][r] + bvv;
        if (do_relu) v = fmaxf(v, 0.0f);
        C[(size_t)(mbase + r) * N + col] = (__bf16)v;
      }
    }
  }
}

// ---------------------------------------------------------------------------
// Fused causal attention for one (b, h, 64-row q tile).
// grid = B*H*(S/64) = 2048 blocks, 256 threads (8 waves).
// Dynamic LDS: scores f32 [64][512] + probs bf16 [64][512] + V^T bf16 [64][520].
// glo/ts terms are constant along k and softmax-invariant -> omitted.
// ---------------------------------------------------------------------------
#define VTSTR 520

__global__ __launch_bounds__(256)
void attn_kernel(const __bf16* __restrict__ Qb, const __bf16* __restrict__ Kb,
                 const __bf16* __restrict__ Vb, float* __restrict__ Ob) {
  extern __shared__ char smem_raw[];
  float*  sc = (float*)smem_raw;                           // 64*512 f32  (128KB)
  __bf16* pb = (__bf16*)(smem_raw + 64 * 512 * 4);         // 64*512 bf16 (64KB)
  __bf16* vt = (__bf16*)(smem_raw + 64 * 512 * 4 + 64 * 512 * 2); // 64*520 bf16
  __shared__ float rsum[64];

  const int blk = blockIdx.x;
  const int qt = blk & 7;
  const int h  = (blk >> 3) & 7;
  const int b  = blk >> 6;
  const int q0 = qt * 64;
  const int kcols = q0 + 64;          // causal column limit (multiple of 64)

  const int tid  = threadIdx.x;
  const int wave = tid >> 5;
  const int lane = tid & 31;
  const int hs   = lane >> 4;
  const int l16  = lane & 15;

  const __bf16* Qh = Qb + (size_t)b * SS * EEc + (size_t)h * DHH;
  const __bf16* Kh = Kb + (size_t)b * SS * EEc + (size_t)h * DHH;
  const __bf16* Vh = Vb + (size_t)b * SS * EEc + (size_t)h * DHH;

  // ---- stage V^T into LDS (coalesced global reads, scattered b16 stores) ----
  {
    const int c0 = (tid & 7) * 8;       // column group 0..56
    for (int kb = (tid >> 3); kb < kcols; kb += 32) {
      union { uint4 q; __bf16 hh[8]; } u;
      u.q = *(const uint4*)(Vh + (size_t)kb * EEc + c0);
#pragma unroll
      for (int i = 0; i < 8; ++i) vt[(c0 + i) * VTSTR + kb] = u.hh[i];
    }
  }

  // ---- scores: S = Q K^T, waves: 4 M-groups x 2 column-frag parities ----
  {
    const int wavem = wave & 3;
    const int wnp   = wave >> 2;
    Frag16 aq[2];
#pragma unroll
    for (int ks = 0; ks < 2; ++ks) {
      const __bf16* qr = Qh + (size_t)(q0 + wavem * 16 + l16) * EEc + ks * 32 + hs * 8;
      aq[ks].q[0] = *(const uint4*)qr;
      aq[ks].q[1] = *(const uint4*)(qr + 16);
    }
    for (int nf = wnp; nf * 16 < kcols; nf += 2) {
      int n0 = nf * 16;
      v8f acc = {};
#pragma unroll
      for (int ks = 0; ks < 2; ++ks) {
        Frag16 bk;
        const __bf16* kr = Kh + (size_t)(n0 + l16) * EEc + ks * 32 + hs * 8;
        bk.q[0] = *(const uint4*)kr;
        bk.q[1] = *(const uint4*)(kr + 16);
        acc = wmma_bf16(aq[ks].v, bk.v, acc);
      }
      int kk = n0 + l16;
#pragma unroll
      for (int r = 0; r < 8; ++r) {
        int qq = wavem * 16 + hs * 8 + r;            // local q row 0..63
        float v = (kk > q0 + qq) ? NEGV : acc[r] * SCALEv;
        sc[qq * 512 + kk] = v;
      }
    }
  }
  __syncthreads();

  // ---- row softmax (4 threads per row); 1/sum deferred to PV epilogue ----
  {
    int row = tid >> 2;
    int sub = tid & 3;
    float mx = -3.0e38f;
    for (int j = sub; j < kcols; j += 4) mx = fmaxf(mx, sc[row * 512 + j]);
#pragma unroll
    for (int o = 1; o < 4; o <<= 1) mx = fmaxf(mx, __shfl_xor(mx, o, 32));
    float sum = 0.0f;
    for (int j = sub; j < kcols; j += 4) {
      float e = __expf(sc[row * 512 + j] - mx);
      pb[row * 512 + j] = (__bf16)e;
      sum += e;
    }
#pragma unroll
    for (int o = 1; o < 4; o <<= 1) sum += __shfl_xor(sum, o, 32);
    if (sub == 0) rsum[row] = 1.0f / sum;
  }
  __syncthreads();

  // ---- PV: O = P V, waves: 4 M-groups x 2 N-halves of 32 ----
  {
    const int wavem = wave >> 1;
    const int waven = wave & 1;
    v8f oacc[2] = {};
    for (int k0 = 0; k0 < kcols; k0 += 32) {
      Frag16 ap;
      const __bf16* pr = pb + (wavem * 16 + l16) * 512 + k0 + hs * 8;
      ap.q[0] = *(const uint4*)pr;
      ap.q[1] = *(const uint4*)(pr + 16);
#pragma unroll
      for (int g = 0; g < 2; ++g) {
        Frag16 bv;
        const __bf16* vr = vt + (size_t)(waven * 32 + g * 16 + l16) * VTSTR + k0 + hs * 8;
        bv.q[0] = *(const uint4*)vr;
        bv.q[1] = *(const uint4*)(vr + 16);
        oacc[g] = wmma_bf16(ap.v, bv.v, oacc[g]);
      }
    }
#pragma unroll
    for (int g = 0; g < 2; ++g) {
      int col = waven * 32 + g * 16 + l16;
#pragma unroll
      for (int r = 0; r < 8; ++r) {
        int qq = wavem * 16 + hs * 8 + r;
        Ob[((size_t)b * SS + q0 + qq) * EEc + (size_t)h * DHH + col] =
            oacc[g][r] * rsum[qq];
      }
    }
  }
}

// ---------------------------------------------------------------------------
// Layer combiners
// ---------------------------------------------------------------------------
__global__ void post_layer0_kernel(const float* __restrict__ attn,
                                   float* __restrict__ acc,
                                   __bf16* __restrict__ kv, int n) {
  int i = blockIdx.x * 256 + threadIdx.x;
  if (i < n) { float v = attn[i]; acc[i] = v; kv[i] = (__bf16)v; }
}
__global__ void post_layer1_kernel(const float* __restrict__ attn,
                                   float* __restrict__ acc, int n) {
  int i = blockIdx.x * 256 + threadIdx.x;
  if (i < n) acc[i] += fmaxf(attn[i], 0.0f);
}

// ---------------------------------------------------------------------------
// Final projection: out[row] = dot(acc[row,:], Wout) + bout.  One wave per row.
// ---------------------------------------------------------------------------
__global__ __launch_bounds__(256)
void out_proj_kernel(const float* __restrict__ acc, const float* __restrict__ Wout,
                     const float* __restrict__ bout, float* __restrict__ out) {
  int row  = blockIdx.x * 8 + (threadIdx.x >> 5);
  int lane = threadIdx.x & 31;
  const float* a = acc + (size_t)row * EEc;
  float s = 0.0f;
  for (int j = lane; j < EEc; j += 32) s += a[j] * Wout[j];
#pragma unroll
  for (int o = 16; o > 0; o >>= 1) s += __shfl_xor(s, o, 32);
  if (lane == 0) out[row] = s + bout[0];
}

// ---------------------------------------------------------------------------
extern "C" void kernel_launch(void* const* d_in, const int* in_sizes, int n_in,
                              void* d_out, int out_size, void* d_ws, size_t ws_size,
                              hipStream_t stream) {
  (void)in_sizes; (void)n_in; (void)out_size; (void)ws_size;
  const int*   item_inputs = (const int*)d_in[0];
  const int*   skill_inputs= (const int*)d_in[1];
  const int*   label_inputs= (const int*)d_in[2];
  const int*   item_ids    = (const int*)d_in[3];
  const int*   skill_ids   = (const int*)d_in[4];
  const float* item_emb    = (const float*)d_in[5];
  const float* skill_emb   = (const float*)d_in[6];
  const float* W_in        = (const float*)d_in[7];
  const float* b_in        = (const float*)d_in[8];
  const float* Wq          = (const float*)d_in[9];
  const float* bq          = (const float*)d_in[10];
  const float* Wk          = (const float*)d_in[11];
  const float* bk          = (const float*)d_in[12];
  const float* Wv          = (const float*)d_in[13];
  const float* bv          = (const float*)d_in[14];
  // d_in[15..21]: Wg/bg/Wtq/btq/Wtk/btk/core — softmax-invariant, unused.
  const float* W_out       = (const float*)d_in[22];
  const float* b_out       = (const float*)d_in[23];
  float* out = (float*)d_out;

  const size_t MROWS = (size_t)BB * SS;          // 16384
  const size_t nE    = MROWS * EEc;              // 8,388,608
  const size_t n2E   = MROWS * 2 * EEc;

  // Workspace carve-up (bytes, 256-aligned)
  char* ws = (char*)d_ws;
  size_t off = 0;
  auto carve = [&](size_t bytes) { char* p = ws + off; off += (bytes + 255) & ~(size_t)255; return p; };
  __bf16* Win_bf = (__bf16*)carve(2 * EEc * EEc * sizeof(__bf16));      // 1024x512
  __bf16* Wq_bf  = (__bf16*)carve((size_t)LLn * EEc * EEc * sizeof(__bf16));
  __bf16* Wk_bf  = (__bf16*)carve((size_t)LLn * EEc * EEc * sizeof(__bf16));
  __bf16* Wv_bf  = (__bf16*)carve((size_t)LLn * EEc * EEc * sizeof(__bf16));
  __bf16* qbf    = (__bf16*)carve(nE  * sizeof(__bf16));
  __bf16* inpbf  = (__bf16*)carve(n2E * sizeof(__bf16));
  __bf16* kvbf   = (__bf16*)carve(nE  * sizeof(__bf16));
  __bf16* Qb     = (__bf16*)carve(nE  * sizeof(__bf16));
  __bf16* Kb     = (__bf16*)carve(nE  * sizeof(__bf16));
  __bf16* Vb     = (__bf16*)carve(nE  * sizeof(__bf16));
  float*  attnf  = (float*)carve(nE * sizeof(float));
  float*  accf   = (float*)carve(nE * sizeof(float));

  // 1) weight conversion fp32 -> bf16
  {
    int n = 2 * EEc * EEc;   // 524288 for each tensor group
    cvt_bf16_kernel<<<dim3((n + 255) / 256), dim3(256), 0, stream>>>(W_in, Win_bf, n);
    cvt_bf16_kernel<<<dim3((n + 255) / 256), dim3(256), 0, stream>>>(Wq,   Wq_bf,  n);
    cvt_bf16_kernel<<<dim3((n + 255) / 256), dim3(256), 0, stream>>>(Wk,   Wk_bf,  n);
    cvt_bf16_kernel<<<dim3((n + 255) / 256), dim3(256), 0, stream>>>(Wv,   Wv_bf,  n);
  }

  // 2) embeddings -> inputs, query
  embed_kernel<<<dim3((unsigned)MROWS), dim3(256), 0, stream>>>(
      item_inputs, skill_inputs, label_inputs, item_ids, skill_ids,
      item_emb, skill_emb, inpbf, qbf);

  // 3) x = relu(inputs @ W_in + b_in) -> kvbf
  gemm_bf16_kernel<<<dim3((unsigned)(MROWS / 128), EEc / 128), dim3(256), 0, stream>>>(
      inpbf, Win_bf, b_in, kvbf, (int)MROWS, EEc, 2 * EEc, 1);

  // 4) layers
  const size_t attn_smem = (size_t)64 * 512 * 4 + (size_t)64 * 512 * 2 +
                           (size_t)64 * VTSTR * 2;   // 257KB dynamic LDS
  for (int l = 0; l < LLn; ++l) {
    const size_t wofs = (size_t)l * EEc * EEc;
    gemm_bf16_kernel<<<dim3((unsigned)(MROWS / 128), EEc / 128), dim3(256), 0, stream>>>(
        qbf,  Wq_bf + wofs, bq + (size_t)l * EEc, Qb, (int)MROWS, EEc, EEc, 0);
    gemm_bf16_kernel<<<dim3((unsigned)(MROWS / 128), EEc / 128), dim3(256), 0, stream>>>(
        kvbf, Wk_bf + wofs, bk + (size_t)l * EEc, Kb, (int)MROWS, EEc, EEc, 0);
    gemm_bf16_kernel<<<dim3((unsigned)(MROWS / 128), EEc / 128), dim3(256), 0, stream>>>(
        kvbf, Wv_bf + wofs, bv + (size_t)l * EEc, Vb, (int)MROWS, EEc, EEc, 0);

    attn_kernel<<<dim3(BB * HH * (SS / 64)), dim3(256), attn_smem, stream>>>(
        Qb, Kb, Vb, attnf);

    if (l == 0) {
      post_layer0_kernel<<<dim3((unsigned)((nE + 255) / 256)), dim3(256), 0, stream>>>(
          attnf, accf, kvbf, (int)nE);
    } else {
      post_layer1_kernel<<<dim3((unsigned)((nE + 255) / 256)), dim3(256), 0, stream>>>(
          attnf, accf, (int)nE);
    }
  }

  // 5) final projection
  out_proj_kernel<<<dim3((unsigned)(MROWS / 8)), dim3(256), 0, stream>>>(
      accf, W_out, b_out, out);
}
